// WindowCrossAttention_79207786873483
// MI455X (gfx1250) — matrix operand order repaired and checked by reading, not compiled, not run on hardware
//
#include <hip/hip_runtime.h>

typedef __attribute__((ext_vector_type(16))) _Float16 v16h;
typedef __attribute__((ext_vector_type(8)))  _Float16 v8h;
typedef __attribute__((ext_vector_type(8)))  float    v8f;

#define DIMC   192
#define NHEAD  6
#define HDIM   32
#define NTOK   49
#define NPAD   64
#define NWIN   4096
#define NMASK  1024
#define XSTR   200   // halfs; 400B rows -> 36*r mod 64 bank walk, conflict-free for 16 rows
#define PSTR   72    // halfs; 144B rows
#define VSTR   72
#define SCALE_Q 0.17677669529663687f  // 32^-0.5

// ---- LDS layout (units: halfs) ----
#define OFF_X1  0                       // 64 x 200  branch_1 tile (f16)
#define OFF_X2  12800                   // 64 x 200  branch_2 tile
#define OFF_Q2  25600                   // 64 x 200  Q (pre-scaled)
#define OFF_K1  38400                   // 64 x 200  K
#define OFF_VT  51200                   // 192 x 72  V transposed (dim-major)
#define OFF_O   65024                   // 64 x 200  attention output (heads concat)
#define OFF_P   77824                   // 6 x 64 x 72  softmax probabilities
#define OFF_BIAS 105472                 // 6*49*49 f32  (TDM destination)
#define OFF_MASK 134284                 // 49*49 f32    (TDM destination)
#define LDSB_BIAS (OFF_BIAS * 2)        // 210944 (bytes)
#define LDSB_MASK (OFF_MASK * 2)        // 268568 (bytes)
#define LDS_BYTES 278176                // < 320KB/WGP

// ---- workspace layout (bytes) ----
#define WS_WQ2  0                       // 192x192 f16, pre-scaled by HD^-0.5
#define WS_WK1  73728
#define WS_WV1  147456
#define WS_WP   221184
#define WS_BQ2  294912                  // 192 f32, pre-scaled
#define WS_BIAS 295680                  // 6*49*49 f32 rel-pos bias
#define WS_END  (295680 + 6*49*49*4)

__device__ __forceinline__ v16h ld_frag(const _Float16* p0, const _Float16* p1) {
  v8h lo = *(const v8h*)p0;
  v8h hi = *(const v8h*)p1;
  return __builtin_shufflevector(lo, hi, 0,1,2,3,4,5,6,7,8,9,10,11,12,13,14,15);
}

__device__ __forceinline__ v8f wmma16(v16h a, v16h b, v8f c) {
  return __builtin_amdgcn_wmma_f32_16x16x32_f16(false, a, false, b, (short)0, c, false, false);
}

// 1-D Tensor-Data-Mover copy: nelem f32 elements global -> LDS (D# per ISA ch.8)
__device__ __forceinline__ void tdm_load_1d(const void* src, unsigned lds_byte,
                                            unsigned nelem) {
  typedef unsigned int u32x4 __attribute__((ext_vector_type(4)));
  typedef int i32x8 __attribute__((ext_vector_type(8)));
  typedef int i32x4 __attribute__((ext_vector_type(4)));
  unsigned long long ga = (unsigned long long)src;
  u32x4 g0;
  g0[0] = 1u;                                                  // count=1 (valid), user mode
  g0[1] = lds_byte;                                            // lds_addr (bytes)
  g0[2] = (unsigned)ga;                                        // global_addr[31:0]
  g0[3] = (unsigned)((ga >> 32) & 0x01FFFFFFu) | (2u << 30);   // global_addr[56:32] | type=2
  i32x8 g1;
  g1[0] = (int)(2u << 16);                                     // data_size = 4 bytes
  g1[1] = (int)((nelem & 0xFFFFu) << 16);                      // tensor_dim0[15:0]
  g1[2] = (int)((nelem >> 16) | (1u << 16));                   // tensor_dim0[31:16], tensor_dim1=1
  g1[3] = (int)((nelem & 0xFFFFu) << 16);                      // tile_dim0
  g1[4] = 1;                                                   // tile_dim1=1, tile_dim2=0
  g1[5] = (int)nelem;                                          // tensor_dim0_stride[31:0]
  g1[6] = 0;
  g1[7] = 0;
  i32x4 z4 = {0, 0, 0, 0};
#if defined(__clang_major__) && (__clang_major__ >= 23)
  i32x8 z8 = {0, 0, 0, 0, 0, 0, 0, 0};
  __builtin_amdgcn_tensor_load_to_lds(g0, g1, z4, z4, z8, 0);
#else
  __builtin_amdgcn_tensor_load_to_lds(g0, g1, z4, z4, 0);
#endif
}

// ---------------- prep: f32 weights -> f16 (+scaling), build bias table ----------------
__global__ void __launch_bounds__(256) prep_kernel(
    const float* __restrict__ qkv_w1, const float* __restrict__ qkv_w2,
    const float* __restrict__ qkv_b2, const float* __restrict__ proj_w1,
    const float* __restrict__ bias_tbl, char* __restrict__ ws)
{
  _Float16* wq2 = (_Float16*)(ws + WS_WQ2);
  _Float16* wk1 = (_Float16*)(ws + WS_WK1);
  _Float16* wv1 = (_Float16*)(ws + WS_WV1);
  _Float16* wp  = (_Float16*)(ws + WS_WP);
  float* bq2    = (float*)(ws + WS_BQ2);
  float* biasg  = (float*)(ws + WS_BIAS);

  const int gid    = blockIdx.x * blockDim.x + threadIdx.x;
  const int stride = gridDim.x * blockDim.x;

  for (int i = gid; i < DIMC * DIMC; i += stride) {
    wq2[i] = (_Float16)(qkv_w2[i] * SCALE_Q);            // q rows of qkv_w2
    wk1[i] = (_Float16)(qkv_w1[DIMC * DIMC + i]);        // k rows of qkv_w1
    wv1[i] = (_Float16)(qkv_w1[2 * DIMC * DIMC + i]);    // v rows of qkv_w1
    wp[i]  = (_Float16)(proj_w1[i]);
  }
  for (int i = gid; i < DIMC; i += stride) bq2[i] = qkv_b2[i] * SCALE_Q;

  for (int t = gid; t < NHEAD * NTOK * NTOK; t += stride) {
    int h   = t / (NTOK * NTOK);
    int rem = t - h * NTOK * NTOK;
    int n = rem / NTOK, m = rem - (rem / NTOK) * NTOK;
    int di = n / 7 - m / 7 + 6;
    int dj = n % 7 - m % 7 + 6;
    biasg[t] = bias_tbl[(di * 13 + dj) * NHEAD + h];
  }
}

// ---------------- fused per-window attention ----------------
__global__ void __launch_bounds__(256) attn_kernel(
    const float* __restrict__ x1, const float* __restrict__ x2,
    const float* __restrict__ maskg, const float* __restrict__ qkv_b1,
    const float* __restrict__ proj_b1, const char* __restrict__ ws,
    float* __restrict__ out)
{
  extern __shared__ _Float16 sm[];
  const _Float16* wq2 = (const _Float16*)(ws + WS_WQ2);
  const _Float16* wk1 = (const _Float16*)(ws + WS_WK1);
  const _Float16* wv1 = (const _Float16*)(ws + WS_WV1);
  const _Float16* wp  = (const _Float16*)(ws + WS_WP);
  const float* bq2    = (const float*)(ws + WS_BQ2);
  const float* biasg  = (const float*)(ws + WS_BIAS);

  const int b    = blockIdx.x;
  const int tid  = threadIdx.x;
  const int wave = tid >> 5;
  const int lane = tid & 31;
  const int lp   = lane & 15;   // D-layout: column; A-layout: row
  const int hi   = lane >> 4;   // half-wave select

  // ---- kick off TDM copies: mask window + rel-pos bias -> LDS (overlaps ph0/ph1) ----
  const int widx = b & (NMASK - 1);
  if (wave == 0) {
    tdm_load_1d(maskg + (size_t)widx * NTOK * NTOK, LDSB_MASK, NTOK * NTOK);
  } else if (wave == 1) {
    tdm_load_1d(biasg, LDSB_BIAS, NHEAD * NTOK * NTOK);
  }

  // warm weights into cache (global_prefetch_b8)
  for (int i = tid; i < WS_BIAS / 1024; i += 256)
    __builtin_prefetch(ws + i * 1024, 0, 2);

  // ---- phase 0: stage X1/X2 into LDS as f16, zero padding rows ----
  {
    const float* s1 = x1 + (size_t)b * NTOK * DIMC;
    const float* s2 = x2 + (size_t)b * NTOK * DIMC;
    for (int i = tid; i < NPAD * XSTR; i += 256) {
      int r = i / XSTR, c = i - r * XSTR;
      bool ok = (r < NTOK) && (c < DIMC);
      float a = ok ? s1[r * DIMC + c] : 0.f;
      float bb = ok ? s2[r * DIMC + c] : 0.f;
      sm[OFF_X1 + i] = (_Float16)a;
      sm[OFF_X2 + i] = (_Float16)bb;
    }
  }
  __syncthreads();

  // ---- phase 1: QKV GEMMs  (3 mats x 4 Mtiles x 12 Ntiles = 144 tile-tasks) ----
  for (int t = wave; t < 144; t += 8) {
    int mat = t / 48;            // 0=Q2 (from X2), 1=K1, 2=V1 (from X1)
    int rem = t - mat * 48;
    int mt  = rem & 3;
    int nt  = rem >> 2;          // 0..11
    const _Float16* A  = sm + (mat == 0 ? OFF_X2 : OFF_X1);
    const _Float16* Wb = (mat == 0) ? wq2 : (mat == 1 ? wk1 : wv1);
    int ncol = nt * 16 + lp;
    float cb = (mat == 0) ? bq2[ncol] : qkv_b1[mat * DIMC + ncol];
    v8f acc;
#pragma unroll
    for (int r = 0; r < 8; ++r) acc[r] = cb;
    const int am = mt * 16 + lp;
#pragma unroll
    for (int k = 0; k < 6; ++k) {
      v16h a = ld_frag(A + am * XSTR + k * 32 + hi * 16,
                       A + am * XSTR + k * 32 + hi * 16 + 8);
      const _Float16* bp = Wb + ncol * DIMC + k * 32 + hi * 16;
      v16h bfr = ld_frag(bp, bp + 8);
      acc = wmma16(a, bfr, acc);
    }
    if (mat == 2) {              // store V transposed: VT[dim][token]
      _Float16* VT = sm + OFF_VT;
#pragma unroll
      for (int r = 0; r < 8; ++r)
        VT[ncol * VSTR + mt * 16 + r + 8 * hi] = (_Float16)acc[r];
    } else {
      _Float16* D = sm + (mat == 0 ? OFF_Q2 : OFF_K1);
#pragma unroll
      for (int r = 0; r < 8; ++r)
        D[(mt * 16 + r + 8 * hi) * XSTR + ncol] = (_Float16)acc[r];
    }
  }
  // TDM data must be visible to all waves past this barrier
  if (wave < 2) __builtin_amdgcn_s_wait_tensorcnt(0);
  __syncthreads();

  // ---- phase 2: scores + bias + mask + softmax -> P (6 heads x 4 Mtiles) ----
  const float* biasLds = (const float*)(sm + OFF_BIAS);
  const float* maskLds = (const float*)(sm + OFF_MASK);
  for (int t = wave; t < 24; t += 8) {
    int h = t >> 2, mt = t & 3;
    const int am = mt * 16 + lp;
    v16h a = ld_frag(sm + OFF_Q2 + am * XSTR + h * HDIM + hi * 16,
                     sm + OFF_Q2 + am * XSTR + h * HDIM + hi * 16 + 8);
    v8f acc[4];
#pragma unroll
    for (int nt = 0; nt < 4; ++nt) {
      int nrow = nt * 16 + lp;   // key token index for this lane's column
      const _Float16* bp = sm + OFF_K1 + nrow * XSTR + h * HDIM + hi * 16;
      v16h bfr = ld_frag(bp, bp + 8);
      v8f c = {};
      acc[nt] = wmma16(a, bfr, c);
    }
    _Float16* P = sm + OFF_P + h * (NPAD * PSTR);
#pragma unroll
    for (int r = 0; r < 8; ++r) {
      int row = mt * 16 + r + 8 * hi;
      float v[4];
      float mx = -1e30f;
#pragma unroll
      for (int nt = 0; nt < 4; ++nt) {
        int col = nt * 16 + lp;
        float s = acc[nt][r];
        if (row < NTOK && col < NTOK)
          s += biasLds[(h * NTOK + row) * NTOK + col] + maskLds[row * NTOK + col];
        if (col >= NTOK) s = -1e30f;
        v[nt] = s;
        mx = fmaxf(mx, s);
      }
      mx = fmaxf(mx, __shfl_xor(mx, 1, 32));
      mx = fmaxf(mx, __shfl_xor(mx, 2, 32));
      mx = fmaxf(mx, __shfl_xor(mx, 4, 32));
      mx = fmaxf(mx, __shfl_xor(mx, 8, 32));
      float sum = 0.f;
#pragma unroll
      for (int nt = 0; nt < 4; ++nt) { v[nt] = __expf(v[nt] - mx); sum += v[nt]; }
      sum += __shfl_xor(sum, 1, 32);
      sum += __shfl_xor(sum, 2, 32);
      sum += __shfl_xor(sum, 4, 32);
      sum += __shfl_xor(sum, 8, 32);
      float inv = 1.0f / sum;
#pragma unroll
      for (int nt = 0; nt < 4; ++nt)
        P[row * PSTR + nt * 16 + lp] = (_Float16)(v[nt] * inv);
    }
  }
  __syncthreads();

  // ---- phase 3: O = P @ V  (6 heads x 4 Mtiles x 2 Ntiles, K=64) ----
  for (int t = wave; t < 48; t += 8) {
    int h = t >> 3, rem = t & 7, mt = rem & 3, nt = rem >> 2;
    const _Float16* P = sm + OFF_P + h * (NPAD * PSTR);
    const int am = mt * 16 + lp;
    v8f acc = {};
#pragma unroll
    for (int kt = 0; kt < 2; ++kt) {
      v16h a = ld_frag(P + am * PSTR + kt * 32 + hi * 16,
                       P + am * PSTR + kt * 32 + hi * 16 + 8);
      int ncol = h * HDIM + nt * 16 + lp;
      const _Float16* bp = sm + OFF_VT + ncol * VSTR + kt * 32 + hi * 16;
      v16h bfr = ld_frag(bp, bp + 8);
      acc = wmma16(a, bfr, acc);
    }
    _Float16* O = sm + OFF_O;
#pragma unroll
    for (int r = 0; r < 8; ++r)
      O[(mt * 16 + r + 8 * hi) * XSTR + h * HDIM + nt * 16 + lp] = (_Float16)acc[r];
  }
  __syncthreads();

  // ---- phase 4: projection, write (o1, o1) ----
  const size_t HALF = (size_t)NWIN * NTOK * DIMC;
  float* ob = out + (size_t)b * NTOK * DIMC;
  for (int t = wave; t < 48; t += 8) {
    int mt = t & 3, nt = t >> 2;  // nt 0..11
    int ncol = nt * 16 + lp;
    float cb = proj_b1[ncol];
    v8f acc;
#pragma unroll
    for (int r = 0; r < 8; ++r) acc[r] = cb;
    const int am = mt * 16 + lp;
#pragma unroll
    for (int k = 0; k < 6; ++k) {
      v16h a = ld_frag(sm + OFF_O + am * XSTR + k * 32 + hi * 16,
                       sm + OFF_O + am * XSTR + k * 32 + hi * 16 + 8);
      const _Float16* bp = wp + ncol * DIMC + k * 32 + hi * 16;
      v16h bfr = ld_frag(bp, bp + 8);
      acc = wmma16(a, bfr, acc);
    }
#pragma unroll
    for (int r = 0; r < 8; ++r) {
      int m = mt * 16 + r + 8 * hi;
      if (m < NTOK) {
        size_t off = (size_t)m * DIMC + ncol;
        ob[off]        = acc[r];
        ob[HALF + off] = acc[r];
      }
    }
  }
}

extern "C" void kernel_launch(void* const* d_in, const int* in_sizes, int n_in,
                              void* d_out, int out_size, void* d_ws, size_t ws_size,
                              hipStream_t stream) {
  (void)in_sizes; (void)n_in; (void)out_size; (void)ws_size;
  const float* x1      = (const float*)d_in[0];
  const float* x2      = (const float*)d_in[1];
  const float* mask    = (const float*)d_in[2];
  const float* qkv_w1  = (const float*)d_in[3];
  const float* qkv_b1  = (const float*)d_in[4];
  const float* qkv_w2  = (const float*)d_in[5];
  const float* qkv_b2  = (const float*)d_in[6];
  const float* proj_w1 = (const float*)d_in[7];
  const float* proj_b1 = (const float*)d_in[8];
  const float* bias_t1 = (const float*)d_in[11];
  char*  ws  = (char*)d_ws;
  float* out = (float*)d_out;

  prep_kernel<<<128, 256, 0, stream>>>(qkv_w1, qkv_w2, qkv_b2, proj_w1, bias_t1, ws);

  hipFuncSetAttribute(reinterpret_cast<const void*>(attn_kernel),
                      hipFuncAttributeMaxDynamicSharedMemorySize, LDS_BYTES);
  attn_kernel<<<NWIN, 256, LDS_BYTES, stream>>>(x1, x2, mask, qkv_b1, proj_b1, ws, out);
}
